// EdgeFormerPatchEmbedding_1743756722826
// MI455X (gfx1250) — matrix operand order repaired and checked by reading, not compiled, not run on hardware
//
#include <hip/hip_runtime.h>

// ---------------------------------------------------------------------------
// EdgeFormer patch embedding on MI455X (gfx1250, wave32, WMMA):
//   conv16x16/s16 == GEMM: A[25088x768] (patches) * B^T[768x768] (weights)
//   + bias, then per-row int4 fake-quant fused in the GEMM epilogue.
// Pipeline: pack_x (im2col+f32->f16), pack_w (f32->f16), gemm_quant (WMMA).
// v4: explicit double-buffered software pipeline over the fully unrolled
//     K-loop (compile-time parity -> buffers stay in registers) plus
//     sched_group_barrier fences (VMEM->DS->WMMA per iteration) so the
//     scheduler cannot sink prefetch loads back under their consumers.
//     WMMAs of iteration i run on data prefetched in iteration i-1.
// ---------------------------------------------------------------------------

typedef _Float16 v8h  __attribute__((ext_vector_type(8)));
typedef _Float16 v16h __attribute__((ext_vector_type(16)));
typedef float    v8f  __attribute__((ext_vector_type(8)));

#define M_TOTAL 25088   // 128 * 196 patches
#define K_DIM   768     // 3 * 16 * 16
#define N_DIM   768     // embed dim
#define K_STEPS 24      // 768 / 32
#define LDS_STRIDE 776  // 768 + 8 halves: 1552 B/row = 388 dw = 4 mod 64 banks

#if defined(__has_builtin)
#if __has_builtin(__builtin_amdgcn_sched_group_barrier)
#define SCHED_GROUP(mask, size) __builtin_amdgcn_sched_group_barrier(mask, size, 0)
#endif
#endif
#ifndef SCHED_GROUP
#define SCHED_GROUP(mask, size)
#endif

// ---------------- pack kernels ----------------

// A[m][k] = x[b, c, ph*16+i, pw*16+j], m = b*196 + ph*14 + pw, k = c*256+i*16+j
// One thread converts 8 contiguous k (j-run of 8) -> one b128 store.
__global__ void __launch_bounds__(256)
pack_x_kernel(const float* __restrict__ x, _Float16* __restrict__ A) {
    int v  = blockIdx.x * blockDim.x + threadIdx.x;   // 0 .. 25088*96-1 (exact)
    int m  = v / 96;
    int k8 = v - m * 96;
    int k  = k8 * 8;
    int c  = k >> 8;            // /256
    int r  = k & 255;
    int i  = r >> 4;
    int j  = r & 15;            // 0 or 8
    int b  = m / 196;
    int p  = m - b * 196;
    int ph = p / 14;
    int pw = p - ph * 14;

    const float* src = x + (((size_t)(b * 3 + c) * 224 + (ph * 16 + i)) * 224
                            + pw * 16 + j);           // 32B aligned
    const float4* s4 = (const float4*)src;
    float4 f0 = s4[0];
    float4 f1 = s4[1];
    v8h h;
    h[0] = (_Float16)f0.x; h[1] = (_Float16)f0.y;
    h[2] = (_Float16)f0.z; h[3] = (_Float16)f0.w;
    h[4] = (_Float16)f1.x; h[5] = (_Float16)f1.y;
    h[6] = (_Float16)f1.z; h[7] = (_Float16)f1.w;
    *(v8h*)(A + (size_t)m * K_DIM + k) = h;
}

// Weights [768][3][16][16] flatten to row-major [N][K] already: pure convert.
__global__ void __launch_bounds__(256)
pack_w_kernel(const float* __restrict__ w, _Float16* __restrict__ B) {
    int v = blockIdx.x * blockDim.x + threadIdx.x;    // 0 .. 73728-1 (exact)
    size_t base = (size_t)v * 8;
    const float4* s4 = (const float4*)(w + base);
    float4 f0 = s4[0];
    float4 f1 = s4[1];
    v8h h;
    h[0] = (_Float16)f0.x; h[1] = (_Float16)f0.y;
    h[2] = (_Float16)f0.z; h[3] = (_Float16)f0.w;
    h[4] = (_Float16)f1.x; h[5] = (_Float16)f1.y;
    h[6] = (_Float16)f1.z; h[7] = (_Float16)f1.w;
    *(v8h*)(B + base) = h;
}

// ---------------- fused WMMA GEMM + bias + int4 fake-quant ----------------
//
// Block: 256 threads = 8 wave32 waves; computes rows [m0, m0+32) x all 768 N.
// Wave w owns N-slice [w*96, w*96+96) = 6 N-tiles; 2 M-tiles -> 12 D tiles.
// A fragment (16x32 f16): lane = row (lane&15), halves of K by
// lane>>4: elems 0..15 = K[(lane>>4)*16 .. (lane>>4)*16+15] (contiguous).
// A frags come from the LDS-staged tile; B frags stream from L2-resident
// global weights: lane = column (lane&15), 16 contiguous K at (lane>>4)*16.
__global__ void __launch_bounds__(256)
gemm_quant_kernel(const _Float16* __restrict__ A, const _Float16* __restrict__ B,
                  const float* __restrict__ bias, float* __restrict__ out) {
    __shared__ _Float16 As[32 * LDS_STRIDE];   // 49,664 B
    __shared__ int rowmax[32];

    const int tid  = threadIdx.x;
    const int lane = tid & 31;
    const int wave = tid >> 5;            // 0..7
    const int m0   = blockIdx.x * 32;
    const int n0   = wave * 96;
    const int arow = lane & 15;
    const int half = lane >> 4;

    if (tid < 32) rowmax[tid] = 0;

    // ---- stage A slice: 32 rows x 768 halves, coalesced b128 global loads,
    //      padded-stride b128 LDS stores (3072 chunks / 256 thr = 12 each)
#pragma unroll
    for (int it = 0; it < 12; ++it) {
        int ci  = tid + it * 256;          // 0..3071
        int row = ci / 96;
        int c8  = ci - row * 96;
        v8h d = *(const v8h*)(A + (size_t)(m0 + row) * K_DIM + c8 * 8);
        *(v8h*)(&As[row * LDS_STRIDE + c8 * 8]) = d;
    }
    __syncthreads();

    v8f acc[2][6];
    const v8f vzero = {0.f, 0.f, 0.f, 0.f, 0.f, 0.f, 0.f, 0.f};
#pragma unroll
    for (int mt = 0; mt < 2; ++mt)
#pragma unroll
        for (int t = 0; t < 6; ++t) acc[mt][t] = vzero;

    const _Float16* a0l = &As[arow * LDS_STRIDE + half * 16];
    const _Float16* a1l = a0l + 16 * LDS_STRIDE;
    const _Float16* bp  = B + (size_t)(n0 + arow) * K_DIM + half * 16;

    auto loadB = [&](int kk, v16h* dst) {
#pragma unroll
        for (int t = 0; t < 6; ++t)
            dst[t] = *(const v16h*)(bp + (size_t)t * 16 * K_DIM + kk);
    };
    auto loadA = [&](int kk, v16h* dst) {
        v8h lo0 = *(const v8h*)(a0l + kk);
        v8h hi0 = *(const v8h*)(a0l + kk + 8);
        v8h lo1 = *(const v8h*)(a1l + kk);
        v8h hi1 = *(const v8h*)(a1l + kk + 8);
        dst[0] = __builtin_shufflevector(lo0, hi0,
                     0,1,2,3,4,5,6,7,8,9,10,11,12,13,14,15);
        dst[1] = __builtin_shufflevector(lo1, hi1,
                     0,1,2,3,4,5,6,7,8,9,10,11,12,13,14,15);
    };

    // ---- software-pipelined K loop: prefetch k+1 while computing k ----
    v16h bufB[2][6];
    v16h bufA[2][2];
    loadB(0, bufB[0]);
    loadA(0, bufA[0]);
#pragma unroll
    for (int ki = 0; ki < K_STEPS; ++ki) {
        const int cur = ki & 1;
        const int nxt = cur ^ 1;
        if (ki < K_STEPS - 1) {            // compile-time after full unroll
            loadB((ki + 1) * 32, bufB[nxt]);
            loadA((ki + 1) * 32, bufA[nxt]);
        }
#pragma unroll
        for (int t = 0; t < 6; ++t) {
            acc[0][t] = __builtin_amdgcn_wmma_f32_16x16x32_f16(
                false, bufA[cur][0], false, bufB[cur][t], (short)0,
                acc[0][t], false, false);
            acc[1][t] = __builtin_amdgcn_wmma_f32_16x16x32_f16(
                false, bufA[cur][1], false, bufB[cur][t], (short)0,
                acc[1][t], false, false);
        }
        // Fence the schedule: next-iter loads first, then this iter's WMMAs.
        SCHED_GROUP(0x020, 12);   // 12 VMEM reads   (B prefetch)
        SCHED_GROUP(0x100, 4);    //  4 DS reads     (A prefetch)
        SCHED_GROUP(0x008, 12);   // 12 WMMA
    }

    // ---- epilogue: bias add ----
#pragma unroll
    for (int mt = 0; mt < 2; ++mt)
#pragma unroll
        for (int t = 0; t < 6; ++t) {
            float bv = bias[n0 + t * 16 + arow];
#pragma unroll
            for (int r = 0; r < 8; ++r) acc[mt][t][r] += bv;
        }

    // ---- per-row |max|: D layout: VGPR r, lane -> M = r + 8*half, N = lane&15
    // shfl_xor over masks 1..8 reduces within each 16-lane half (wave32).
#pragma unroll
    for (int mt = 0; mt < 2; ++mt)
#pragma unroll
        for (int r = 0; r < 8; ++r) {
            float mx = 0.f;
#pragma unroll
            for (int t = 0; t < 6; ++t) mx = fmaxf(mx, fabsf(acc[mt][t][r]));
            mx = fmaxf(mx, __shfl_xor(mx, 1, 32));
            mx = fmaxf(mx, __shfl_xor(mx, 2, 32));
            mx = fmaxf(mx, __shfl_xor(mx, 4, 32));
            mx = fmaxf(mx, __shfl_xor(mx, 8, 32));
            if (arow == 0)  // non-negative floats: int max == float max
                atomicMax(&rowmax[mt * 16 + r + 8 * half], __float_as_int(mx));
        }
    __syncthreads();

    // ---- int4 fake-quant + store ----
#pragma unroll
    for (int mt = 0; mt < 2; ++mt)
#pragma unroll
        for (int r = 0; r < 8; ++r) {
            int row  = mt * 16 + r + 8 * half;
            float s  = __int_as_float(rowmax[row]) * (1.f / 7.f);
            if (s == 0.f) s = 1.f;
            float inv = 1.f / s;
            size_t orow = (size_t)(m0 + row) * N_DIM + n0 + arow;
#pragma unroll
            for (int t = 0; t < 6; ++t) {
                float q = rintf(acc[mt][t][r] * inv);
                q = fminf(7.f, fmaxf(-8.f, q));
                out[orow + t * 16] = q * s;
            }
        }
}

// ---------------- launch ----------------

extern "C" void kernel_launch(void* const* d_in, const int* in_sizes, int n_in,
                              void* d_out, int out_size, void* d_ws, size_t ws_size,
                              hipStream_t stream) {
    const float* x    = (const float*)d_in[0];   // (128,3,224,224)
    const float* w    = (const float*)d_in[1];   // (768,3,16,16)
    const float* bias = (const float*)d_in[2];   // (768,)
    float* out = (float*)d_out;                  // (128,196,768) f32

    _Float16* Ah = (_Float16*)d_ws;                          // 25088*768 f16
    _Float16* Bh = Ah + (size_t)M_TOTAL * K_DIM;             // 768*768  f16

    // pack: 25088*96 = 2,408,448 vec8 threads; 73728 vec8 threads for weights
    pack_x_kernel<<<(M_TOTAL * 96) / 256, 256, 0, stream>>>(x, Ah);
    pack_w_kernel<<<(N_DIM * K_DIM / 8) / 256, 256, 0, stream>>>(w, Bh);

    // GEMM: 25088/32 = 784 blocks of 8 wave32 waves, full-N per block
    gemm_quant_kernel<<<M_TOTAL / 32, 256, 0, stream>>>(Ah, Bh, bias, out);
}
